// YOLONano_58368605553305
// MI455X (gfx1250) — compile-verified, matrix operand order checked by reading, not run on hardware
//
#include <hip/hip_runtime.h>

// ---------------------------------------------------------------------------
// YOLO post-process for MI455X (gfx1250, wave32).
//   Kernel 1: decode + softmax via V_WMMA_F32_16X16X32_F16 (ones-matrix rowsum)
//   Kernel 2: O(N^2) stable descending rank-sort, LDS staging via
//             global_load_async_to_lds_b32 + s_wait_asynccnt (CDNA5 async path)
//   Kernel 3: faithful greedy NMS, single WGP, LDS suppression bitmask
// ---------------------------------------------------------------------------

#define N_BOX 25200
#define N_S   19200   // level-0 boxes (80x80x3)
#define N_M   24000   // + level-1 boxes (40x40x3)
#define SUPP_WORDS ((N_BOX + 31) / 32)

typedef __attribute__((ext_vector_type(16))) _Float16 v16h;
typedef __attribute__((ext_vector_type(8)))  float    v8f;

__constant__ float c_anch[9][2] = {
    {12.f, 16.f}, {19.f, 36.f}, {40.f, 28.f},
    {36.f, 75.f}, {76.f, 55.f}, {72.f, 146.f},
    {142.f, 110.f}, {192.f, 243.f}, {459.f, 401.f}};

__device__ __forceinline__ float sigmoidf_(float x) {
    return 1.f / (1.f + __expf(-x));
}

// ---------------------------------------------------------------------------
// Kernel 1: one wave32 per 16 boxes. Lanes (2b, 2b+1) own box b (40 classes
// each). Sum of exp(cls - cmax) over the 80 classes is computed with three
// V_WMMA_F32_16X16X32_F16 ops against an all-ones B matrix (row-sum).
// ---------------------------------------------------------------------------
__global__ __launch_bounds__(32) void yolo_decode_wmma(
    const float* __restrict__ ps, const float* __restrict__ pm,
    const float* __restrict__ pl, float* __restrict__ out,
    float* __restrict__ nbox_ws, float* __restrict__ conf_ws)
{
    __shared__ _Float16 tile[16][96];   // 16 boxes x (80 classes + 16 zero pad)
    __shared__ float    sums[16];

    const int lane = threadIdx.x;          // 0..31
    const int boxL = lane >> 1;            // 0..15
    const int half = lane & 1;             // 0/1 -> classes [0,40) / [40,80)
    const int g    = blockIdx.x * 16 + boxL;

    const float* base; int HW, F, li, lvl; float stride;
    if (g < N_S)      { base = ps; HW = 6400; F = 80; stride = 8.f;  li = g;       lvl = 0; }
    else if (g < N_M) { base = pm; HW = 1600; F = 40; stride = 16.f; li = g - N_S; lvl = 1; }
    else              { base = pl; HW = 400;  F = 20; stride = 32.f; li = g - N_M; lvl = 2; }
    const int pos = li / 3;
    const int a   = li - pos * 3;

    // ---- class logits: 40 per lane, strided by HW (coalesced across lanes) --
    float lg[40];
    const float* cp = base + (size_t)(3 + a * 80 + half * 40) * HW + pos;
    float cmax = -3.4e38f; int carg = 0;
    #pragma unroll
    for (int k = 0; k < 40; ++k) {
        float v = cp[(size_t)k * HW];
        lg[k] = v;
        if (v > cmax) { cmax = v; carg = half * 40 + k; }
    }
    // combine the two halves of each box with a wave32 xor-shuffle
    float om = __shfl_xor(cmax, 1, 32);
    int   oa = __shfl_xor(carg, 1, 32);
    if (om > cmax || (om == cmax && oa < carg)) { cmax = om; carg = oa; }

    // exp(cls - cmax) -> f16 tile; zero the 16 pad columns
    #pragma unroll
    for (int k = 0; k < 40; ++k)
        tile[boxL][half * 40 + k] = (_Float16)__expf(lg[k] - cmax);
    #pragma unroll
    for (int k = 0; k < 8; ++k)
        tile[boxL][80 + half * 8 + k] = (_Float16)0.f;
    __syncthreads();

    // ---- row sums via WMMA: D = A(16x96, 3 chunks of K=32) x ones(32x16) ---
    const int M  = lane & 15;
    const int hi = lane >> 4;
    v16h B;
    #pragma unroll
    for (int i = 0; i < 16; ++i) B[i] = (_Float16)1.0f;
    v8f acc = {};
    #pragma unroll
    for (int t = 0; t < 3; ++t) {
        v16h A;
        #pragma unroll
        for (int i = 0; i < 8; ++i) A[i]     = tile[M][t * 32 + hi * 8 + i];
        #pragma unroll
        for (int i = 0; i < 8; ++i) A[8 + i] = tile[M][t * 32 + 16 + hi * 8 + i];
        acc = __builtin_amdgcn_wmma_f32_16x16x32_f16(
            /*neg_a=*/false, A, /*neg_b=*/false, B,
            /*c_mod=*/(short)0, acc, /*reuse_a=*/false, /*reuse_b=*/false);
    }
    if (M == 0) {                 // lanes 0 & 16 hold column N=0 of D
        #pragma unroll
        for (int r = 0; r < 8; ++r) sums[hi * 8 + r] = acc[r];
    }
    __syncthreads();

    // ---- per-box finalize on even lanes ------------------------------------
    if (half == 0) {
        const float sumexp = sums[boxL];
        const float objv   = base[(size_t)a * HW + pos];
        const float* rp    = base + (size_t)(243 + a * 4) * HW + pos;
        const float rx = rp[0], ry = rp[(size_t)HW],
                    rw = rp[(size_t)2 * HW], rh = rp[(size_t)3 * HW];
        const float gx = (float)(pos % F), gy = (float)(pos / F);
        const float cx = (sigmoidf_(rx) + gx) * stride;
        const float cy = (sigmoidf_(ry) + gy) * stride;
        const float bw = __expf(rw) * c_anch[lvl * 3 + a][0];
        const float bh = __expf(rh) * c_anch[lvl * 3 + a][1];
        const float inv = 1.f / 640.f;
        const float x1 = fminf(fmaxf((cx - 0.5f * bw) * inv, 0.f), 1.f);
        const float y1 = fminf(fmaxf((cy - 0.5f * bh) * inv, 0.f), 1.f);
        const float x2 = fminf(fmaxf((cx + 0.5f * bw) * inv, 0.f), 1.f);
        const float y2 = fminf(fmaxf((cy + 0.5f * bh) * inv, 0.f), 1.f);
        const float conf = sigmoidf_(objv) / sumexp;   // softmax max = 1/sumexp

        out[g * 4 + 0] = x1; out[g * 4 + 1] = y1;
        out[g * 4 + 2] = x2; out[g * 4 + 3] = y2;
        out[4 * N_BOX + g] = conf;
        out[5 * N_BOX + g] = (float)carg;

        const float off = 4.f * (float)carg;           // class-offset NMS trick
        nbox_ws[g * 4 + 0] = x1 + off; nbox_ws[g * 4 + 1] = y1 + off;
        nbox_ws[g * 4 + 2] = x2 + off; nbox_ws[g * 4 + 3] = y2 + off;
        conf_ws[g] = conf;
    }
}

// ---------------------------------------------------------------------------
// Kernel 2: stable descending argsort via O(N^2) rank counting. conf tiles
// are staged into LDS with the CDNA5 async-to-LDS path (ASYNCcnt).
// ---------------------------------------------------------------------------
__global__ __launch_bounds__(256) void yolo_rank_scatter(
    const float* __restrict__ conf, const float* __restrict__ nbox,
    float4* __restrict__ sbox, float* __restrict__ sarea,
    int* __restrict__ svalid, int* __restrict__ order)
{
    __shared__ float cbuf[256];
    const int tid = threadIdx.x;
    const int i   = blockIdx.x * 256 + tid;
    const float ci = (i < N_BOX) ? conf[i] : 0.f;
    int rank = 0;

    for (int jb = 0; jb < N_BOX; jb += 256) {
        if (jb + 256 <= N_BOX) {
            // async global->LDS copy (GVS): lds_off VGPR, byte-off VGPR, SGPR64 base
            unsigned ldsoff = (unsigned)(uintptr_t)&cbuf[tid];
            unsigned goff   = (unsigned)((jb + tid) * 4);
            asm volatile("global_load_async_to_lds_b32 %0, %1, %2"
                         :: "v"(ldsoff), "v"(goff), "s"(conf) : "memory");
            asm volatile("s_wait_asynccnt 0" ::: "memory");
        } else {
            const int j = jb + tid;
            cbuf[tid] = (j < N_BOX) ? conf[j] : -1.f;  // pad never ties (conf>0)
        }
        __syncthreads();
        if (i < N_BOX) {
            #pragma unroll 8
            for (int k = 0; k < 256; ++k) {
                const float cj = cbuf[k];
                const int   j  = jb + k;
                rank += (cj > ci) || (cj == ci && j < i);
            }
        }
        __syncthreads();
    }

    if (i < N_BOX) {
        const float x1 = nbox[i * 4 + 0], y1 = nbox[i * 4 + 1];
        const float x2 = nbox[i * 4 + 2], y2 = nbox[i * 4 + 3];
        order[rank]  = i;
        sbox[rank]   = make_float4(x1, y1, x2, y2);
        sarea[rank]  = (x2 - x1) * (y2 - y1);
        svalid[rank] = (ci >= 0.001f) ? 1 : 0;
    }
}

// ---------------------------------------------------------------------------
// Kernel 3: faithful sequential greedy NMS. One workgroup (serial dependency),
// LDS suppression bitmask, 32 waves sweep IoUs per step.
// ---------------------------------------------------------------------------
__global__ __launch_bounds__(1024, 1) void yolo_greedy_nms(
    const float4* __restrict__ sbox, const float* __restrict__ sarea,
    const int* __restrict__ svalid, const int* __restrict__ order,
    float* __restrict__ keep_out)
{
    __shared__ unsigned supp[SUPP_WORDS];
    const int tid = threadIdx.x;
    for (int w = tid; w < SUPP_WORDS; w += blockDim.x) supp[w] = 0u;
    __syncthreads();

    for (int i = 0; i < N_BOX; ++i) {
        const bool sup  = (supp[i >> 5] >> (i & 31)) & 1u;
        const bool keep = (svalid[i] != 0) && !sup;
        if (tid == 0) keep_out[order[i]] = keep ? 1.f : 0.f;
        if (keep) {
            const float4 bi = sbox[i];
            const float  ai = sarea[i];
            for (int j = i + 1 + tid; j < N_BOX; j += blockDim.x) {
                if ((supp[j >> 5] >> (j & 31)) & 1u) continue;  // benign race
                const float4 bj = sbox[j];
                const float xx1 = fmaxf(bi.x, bj.x), yy1 = fmaxf(bi.y, bj.y);
                const float xx2 = fminf(bi.z, bj.z), yy2 = fminf(bi.w, bj.w);
                const float wd = fmaxf(1e-28f, xx2 - xx1);
                const float hd = fmaxf(1e-28f, yy2 - yy1);
                const float inter = wd * hd;
                const float ovr = inter / (ai + sarea[j] - inter + 1e-14f);
                if (ovr > 0.6f) atomicOr(&supp[j >> 5], 1u << (j & 31));
            }
        }
        __syncthreads();
    }
}

// ---------------------------------------------------------------------------
extern "C" void kernel_launch(void* const* d_in, const int* in_sizes, int n_in,
                              void* d_out, int out_size, void* d_ws, size_t ws_size,
                              hipStream_t stream)
{
    (void)in_sizes; (void)n_in; (void)out_size; (void)ws_size;
    const float* ps = (const float*)d_in[0];
    const float* pm = (const float*)d_in[1];
    const float* pl = (const float*)d_in[2];
    float* out = (float*)d_out;

    // workspace layout (floats): nbox[4N] | conf[N] | sbox[4N] | sarea[N] |
    //                            svalid[N] | order[N]   (total 12N*4 = 1.21 MB)
    float*  wsf     = (float*)d_ws;
    float*  nbox_ws = wsf;
    float*  conf_ws = wsf + 4 * N_BOX;
    float4* sbox    = (float4*)(wsf + 5 * N_BOX);   // byte offset 504000, 16B aligned
    float*  sarea   = wsf + 9 * N_BOX;
    int*    svalid  = (int*)(wsf + 10 * N_BOX);
    int*    order   = (int*)(wsf + 11 * N_BOX);

    yolo_decode_wmma<<<N_BOX / 16, 32, 0, stream>>>(ps, pm, pl, out, nbox_ws, conf_ws);
    yolo_rank_scatter<<<(N_BOX + 255) / 256, 256, 0, stream>>>(
        conf_ws, nbox_ws, sbox, sarea, svalid, order);
    yolo_greedy_nms<<<1, 1024, 0, stream>>>(sbox, sarea, svalid, order, out + 6 * N_BOX);
}
